// TransformerDiscrete_69054484185517
// MI455X (gfx1250) — matrix operand order repaired and checked by reading, not compiled, not run on hardware
//
#include <hip/hip_runtime.h>

// ---------------- problem constants (match reference) ----------------
#define B_  16
#define N_  641
#define H_  512
#define F_  2048
#define NH_ 8
#define HD_ 64
#define V_  1024
#define NL_ 6
#define M_  (B_ * N_)        // 10256 token rows

typedef unsigned short u16;
typedef __attribute__((ext_vector_type(16))) __bf16         v16bf;
typedef __attribute__((ext_vector_type(8)))  float          v8f;
typedef __attribute__((ext_vector_type(8)))  unsigned short v8u16;

// ---------------- helpers ----------------
__device__ __forceinline__ u16 f2bf(float f) {
  unsigned u = __float_as_uint(f);
  u += 0x7FFFu + ((u >> 16) & 1u);            // round-to-nearest-even
  return (u16)(u >> 16);
}

// Build a v16bf WMMA fragment from two 16-byte chunks (global or LDS).
__device__ __forceinline__ v16bf load_frag(const u16* lo, const u16* hi) {
  union { v16bf v; v8u16 h[2]; } u;
  u.h[0] = *(const v8u16*)lo;
  u.h[1] = *(const v8u16*)hi;
  return u.v;
}

__device__ __forceinline__ v8f wmma_bf16(v16bf a, v16bf b, v8f c) {
  // (neg_a, A, neg_b, B, c_mod, C, reuse_a, reuse_b)
  return __builtin_amdgcn_wmma_f32_16x16x32_bf16(false, a, false, b, (short)0, c,
                                                 false, false);
}

// Pseudo-causal mask, computed analytically (True = disallowed).
// disallowed(q,k) = (k>q) unless key (k-1) is a sentence-0 token of q's paragraph.
__device__ __forceinline__ bool is_masked(int q, int k) {
  if (q >= N_ || k >= N_) return true;
  if (k <= q) return false;
  if (q < N_ - 1) {                 // q indexes into the L=640 region
    int kp = k - 1;                 // k>=1 guaranteed since k>q>=0
    if ((q / 40) == (kp / 40) && (kp % 40) < 8) return false;  // S*T=40, T=8
  }
  return true;
}

// ---------------- embedding + positional encoding ----------------
__global__ void embed_kernel(const int* __restrict__ tok, const float* __restrict__ emb,
                             const float* __restrict__ pe, float* __restrict__ x,
                             u16* __restrict__ xb) {
  int idx = blockIdx.x * blockDim.x + threadIdx.x;
  if (idx >= M_ * H_) return;
  int row = idx / H_, col = idx - row * H_;
  int n = row % N_;
  int t = tok[row];
  float v = emb[(size_t)t * H_ + col] + pe[(size_t)n * H_ + col];
  x[idx]  = v;
  xb[idx] = f2bf(v);
}

// ---------------- fp32 -> bf16 weight conversion ----------------
__global__ void f2bf_kernel(const float* __restrict__ src, u16* __restrict__ dst, int n) {
  int i = blockIdx.x * blockDim.x + threadIdx.x;
  if (i < n) dst[i] = f2bf(src[i]);
}

// ---------------- residual + LayerNorm (wave per row) ----------------
__global__ __launch_bounds__(256) void addln_kernel(
    const float* __restrict__ x, const float* __restrict__ d,
    const float* __restrict__ g, const float* __restrict__ bb,
    float* __restrict__ yf, u16* __restrict__ yb) {
  const int row = blockIdx.x * 8 + (threadIdx.x >> 5);
  if (row >= M_) return;
  const int lane = threadIdx.x & 31;
  const float* xr = x + (size_t)row * H_;
  const float* dr = d ? d + (size_t)row * H_ : nullptr;
  float vals[16];
  float s = 0.f;
#pragma unroll
  for (int i = 0; i < 16; ++i) {
    int c = lane + 32 * i;
    float v = xr[c] + (dr ? dr[c] : 0.f);
    vals[i] = v; s += v;
  }
#pragma unroll
  for (int m = 16; m >= 1; m >>= 1) s += __shfl_xor(s, m, 32);
  const float mean = s * (1.f / H_);
  float vs = 0.f;
#pragma unroll
  for (int i = 0; i < 16; ++i) { float t = vals[i] - mean; vs += t * t; }
#pragma unroll
  for (int m = 16; m >= 1; m >>= 1) vs += __shfl_xor(vs, m, 32);
  const float inv = rsqrtf(vs * (1.f / H_) + 1e-5f);
#pragma unroll
  for (int i = 0; i < 16; ++i) {
    int c = lane + 32 * i;
    float yv = (vals[i] - mean) * inv * g[c] + bb[c];
    yf[(size_t)row * H_ + c] = yv;
    yb[(size_t)row * H_ + c] = f2bf(yv);
  }
}

// ---------------- generic WMMA GEMM: C = A[M,K] * W[N,K]^T + bias ----------------
// 4 waves/block arranged 2x2; wave tile 64(M) x 64(N) = 4x4 WMMA 16x16x32 subtiles.
// 16 WMMAs per 16 B128 loads per K-step; A-frags stay resident, B-frags stream.
__global__ __launch_bounds__(128) void gemm_bf16_kernel(
    const u16* __restrict__ A, const u16* __restrict__ W,
    const float* __restrict__ bias, float* __restrict__ outF,
    u16* __restrict__ outB, int Mdim, int Kdim, int Ndim, int act) {
  const int lane = threadIdx.x & 31;
  const int w    = threadIdx.x >> 5;
  const int kh   = lane >> 4;       // K-half select per WMMA layout
  const int l16  = lane & 15;
  const int m0 = blockIdx.x * 128 + (w >> 1) * 64;
  const int n0 = blockIdx.y * 128 + (w & 1) * 64;

  const u16* Ap[4];
#pragma unroll
  for (int i = 0; i < 4; ++i) {
    int r = m0 + i * 16 + l16;
    if (r >= Mdim) r = Mdim - 1;                 // clamp; masked at store
    Ap[i] = A + (size_t)r * Kdim + kh * 8;
  }
  const u16* Bp[4];
#pragma unroll
  for (int j = 0; j < 4; ++j) {
    int r = n0 + j * 16 + l16;                   // Ndim multiple of 128
    Bp[j] = W + (size_t)r * Kdim + kh * 8;
  }

  v8f acc[4][4];
#pragma unroll
  for (int i = 0; i < 4; ++i)
#pragma unroll
    for (int j = 0; j < 4; ++j)
#pragma unroll
      for (int v = 0; v < 8; ++v) acc[i][j][v] = 0.f;

  for (int k = 0; k < Kdim; k += 32) {
    v16bf af[4];
#pragma unroll
    for (int i = 0; i < 4; ++i) af[i] = load_frag(Ap[i] + k, Ap[i] + k + 16);
#pragma unroll
    for (int j = 0; j < 4; ++j) {
      v16bf bfrag = load_frag(Bp[j] + k, Bp[j] + k + 16);
#pragma unroll
      for (int i = 0; i < 4; ++i)
        acc[i][j] = wmma_bf16(af[i], bfrag, acc[i][j]);
    }
  }

#pragma unroll
  for (int j = 0; j < 4; ++j) {
    const int col = n0 + j * 16 + l16;
    const float bv = bias ? bias[col] : 0.f;
#pragma unroll
    for (int i = 0; i < 4; ++i) {
#pragma unroll
      for (int v = 0; v < 8; ++v) {
        const int row = m0 + i * 16 + v + kh * 8;   // C layout: m = v + 8*(lane/16)
        if (row < Mdim) {
          float xv = acc[i][j][v] + bv;
          if (act == 1)  // exact GELU
            xv = 0.5f * xv * (1.f + erff(xv * 0.70710678118654752f));
          if (outF) outF[(size_t)row * Ndim + col] = xv;
          if (outB) outB[(size_t)row * Ndim + col] = f2bf(xv);
        }
      }
    }
  }
}

// ---------------- flash-style WMMA attention ----------------
// grid = (41 q-blocks, B*NH); one wave per block; qkv packed [M, 3H] bf16.
__global__ __launch_bounds__(32) void attention_kernel(const u16* __restrict__ qkv,
                                                       u16* __restrict__ attout) {
  __shared__ __align__(16) u16 Plds[16 * 32];   // probabilities, A-layout staging
  __shared__ __align__(16) u16 Vt[64 * 32];     // V tile transposed: [hd_col][k']
  const int qb   = blockIdx.x;
  const int bh   = blockIdx.y;
  const int b    = bh >> 3, h = bh & 7;
  const int lane = threadIdx.x;
  const int kh   = lane >> 4, l16 = lane & 15;

  int qr = qb * 16 + l16; if (qr > N_ - 1) qr = N_ - 1;
  const u16* qp = qkv + ((size_t)(b * N_ + qr)) * (3 * H_) + h * HD_;
  const v16bf q0 = load_frag(qp + kh * 8,      qp + 16 + kh * 8);
  const v16bf q1 = load_frag(qp + 32 + kh * 8, qp + 48 + kh * 8);

  float mrow[8], lrow[8];
  v8f O[4];
#pragma unroll
  for (int v = 0; v < 8; ++v) { mrow[v] = -3e38f; lrow[v] = 0.f; }
#pragma unroll
  for (int c = 0; c < 4; ++c)
#pragma unroll
    for (int v = 0; v < 8; ++v) O[c][v] = 0.f;

  // pseudo-causal bound: any allowed key satisfies k <= q+8
  int kbmax = (qb * 16 + 15 + 8) / 32 + 1;
  if (kbmax > 21) kbmax = 21;

  for (int kb = 0; kb < kbmax; ++kb) {
    const int k0 = kb * 32;
    { // stage V tile transposed into LDS (one key row per lane)
      int kr = k0 + lane; if (kr > N_ - 1) kr = N_ - 1;
      const u16* vp = qkv + ((size_t)(b * N_ + kr)) * (3 * H_) + 2 * H_ + h * HD_;
#pragma unroll
      for (int i = 0; i < 8; ++i) {
        v8u16 tv = *(const v8u16*)(vp + i * 8);
#pragma unroll
        for (int j = 0; j < 8; ++j) Vt[(i * 8 + j) * 32 + lane] = tv[j];
      }
    }
    // two 16x16 score tiles: S = Q * K^T  (K-dim = 64 -> 2 WMMAs each)
    v8f S[2];
#pragma unroll
    for (int t = 0; t < 2; ++t) {
      int kr = k0 + t * 16 + l16; if (kr > N_ - 1) kr = N_ - 1;
      const u16* kp = qkv + ((size_t)(b * N_ + kr)) * (3 * H_) + H_ + h * HD_;
      v16bf kf0 = load_frag(kp + kh * 8,      kp + kh * 8 + 16);
      v16bf kf1 = load_frag(kp + 32 + kh * 8, kp + 32 + kh * 8 + 16);
      v8f s;
#pragma unroll
      for (int v = 0; v < 8; ++v) s[v] = 0.f;
      s = wmma_bf16(q0, kf0, s);
      s = wmma_bf16(q1, kf1, s);
      S[t] = s;
    }
    // scale + mask + online softmax (row = v + 8*(lane/16), col = lane%16)
#pragma unroll
    for (int v = 0; v < 8; ++v) {
      const int qi = qb * 16 + v + kh * 8;
#pragma unroll
      for (int t = 0; t < 2; ++t) {
        const int ki = k0 + t * 16 + l16;
        float s = S[t][v] * 0.125f;                // 1/sqrt(HD)
        if (is_masked(qi, ki)) s = -1e9f;          // exact NEG of reference
        S[t][v] = s;
      }
      float tm = fmaxf(S[0][v], S[1][v]);
      tm = fmaxf(tm, __shfl_xor(tm, 1, 32));
      tm = fmaxf(tm, __shfl_xor(tm, 2, 32));
      tm = fmaxf(tm, __shfl_xor(tm, 4, 32));
      tm = fmaxf(tm, __shfl_xor(tm, 8, 32));
      const float nm   = fmaxf(mrow[v], tm);
      const float corr = __expf(mrow[v] - nm);
      mrow[v] = nm;
      const float p0 = __expf(S[0][v] - nm);
      const float p1 = __expf(S[1][v] - nm);
      float rs = p0 + p1;
      rs += __shfl_xor(rs, 1, 32);
      rs += __shfl_xor(rs, 2, 32);
      rs += __shfl_xor(rs, 4, 32);
      rs += __shfl_xor(rs, 8, 32);
      lrow[v] = lrow[v] * corr + rs;
#pragma unroll
      for (int c = 0; c < 4; ++c) O[c][v] *= corr;
      Plds[(v + kh * 8) * 32 + l16]      = f2bf(p0);
      Plds[(v + kh * 8) * 32 + 16 + l16] = f2bf(p1);
    }
    __syncthreads();   // P/Vt stores visible before WMMA operand reload
    const v16bf pf = load_frag(&Plds[l16 * 32 + kh * 8],
                               &Plds[l16 * 32 + 16 + kh * 8]);
#pragma unroll
    for (int c = 0; c < 4; ++c) {
      v16bf vf = load_frag(&Vt[(c * 16 + l16) * 32 + kh * 8],
                           &Vt[(c * 16 + l16) * 32 + kh * 8 + 16]);
      O[c] = wmma_bf16(pf, vf, O[c]);
    }
    __syncthreads();   // consume before next iteration overwrites LDS
  }
  // normalize + store bf16 attention output [M, H]
#pragma unroll
  for (int v = 0; v < 8; ++v) {
    const int qi = qb * 16 + v + kh * 8;
    if (qi < N_) {
      const float invl = 1.f / lrow[v];
#pragma unroll
      for (int c = 0; c < 4; ++c)
        attout[((size_t)(b * N_ + qi)) * H_ + h * HD_ + c * 16 + l16] =
            f2bf(O[c][v] * invl);
    }
  }
}

// ---------------- host orchestration ----------------
extern "C" void kernel_launch(void* const* d_in, const int* in_sizes, int n_in,
                              void* d_out, int out_size, void* d_ws, size_t ws_size,
                              hipStream_t stream) {
  (void)in_sizes; (void)n_in; (void)out_size; (void)ws_size;
  const int*   tokens = (const int*)  d_in[0];
  // d_in[1] = attn_mask (computed analytically on-device instead)
  const float* pe     = (const float*)d_in[2];
  const float* emb    = (const float*)d_in[3];
  const float* qkv_w  = (const float*)d_in[4];
  const float* qkv_b  = (const float*)d_in[5];
  const float* out_w  = (const float*)d_in[6];
  const float* out_b  = (const float*)d_in[7];
  const float* ln1_g  = (const float*)d_in[8];
  const float* ln1_b  = (const float*)d_in[9];
  const float* fc1_w  = (const float*)d_in[10];
  const float* fc1_b  = (const float*)d_in[11];
  const float* fc2_w  = (const float*)d_in[12];
  const float* fc2_b  = (const float*)d_in[13];
  const float* ln2_g  = (const float*)d_in[14];
  const float* ln2_b  = (const float*)d_in[15];
  const float* lnf_g  = (const float*)d_in[16];
  const float* lnf_b  = (const float*)d_in[17];
  const float* dep_w  = (const float*)d_in[18];
  const float* dep_b  = (const float*)d_in[19];
  float* logits = (float*)d_out;

  size_t off = 0;
  auto carve = [&](size_t bytes) {
    char* p = (char*)d_ws + off;
    off += (bytes + 255) & ~(size_t)255;
    return (void*)p;
  };
  float* x    = (float*)carve((size_t)M_ * H_ * 4);       // fp32 residual stream
  u16*   xb   = (u16*)  carve((size_t)M_ * H_ * 2);       // bf16 activations
  u16*   qkvb = (u16*)  carve((size_t)M_ * 3 * H_ * 2);   // bf16 QKV
  u16*   attb = (u16*)  carve((size_t)M_ * H_ * 2);       // bf16 attn out
  float* tmp  = (float*)carve((size_t)M_ * H_ * 4);       // fp32 proj/ffn out
  u16*   hb   = (u16*)  carve((size_t)M_ * F_ * 2);       // bf16 ffn hidden
  u16*   wbuf = (u16*)  carve((size_t)F_ * H_ * 2);       // bf16 weight slice

  auto conv = [&](const float* s, u16* d, int n) {
    f2bf_kernel<<<(n + 255) / 256, 256, 0, stream>>>(s, d, n);
  };
  auto gemm = [&](const u16* A, const u16* W, const float* bias, float* oF, u16* oB,
                  int Md, int Kd, int Nd, int act) {
    dim3 g((Md + 127) / 128, Nd / 128);
    gemm_bf16_kernel<<<g, 128, 0, stream>>>(A, W, bias, oF, oB, Md, Kd, Nd, act);
  };
  auto addln = [&](const float* xi, const float* di, const float* g, const float* b) {
    addln_kernel<<<(M_ + 7) / 8, 256, 0, stream>>>(xi, di, g, b, x, xb);
  };

  embed_kernel<<<(M_ * H_ + 255) / 256, 256, 0, stream>>>(tokens, emb, pe, x, xb);

  for (int l = 0; l < NL_; ++l) {
    conv(qkv_w + (size_t)l * 3 * H_ * H_, wbuf, 3 * H_ * H_);
    gemm(xb, wbuf, qkv_b + (size_t)l * 3 * H_, nullptr, qkvb, M_, H_, 3 * H_, 0);

    attention_kernel<<<dim3(41, B_ * NH_), 32, 0, stream>>>(qkvb, attb);

    conv(out_w + (size_t)l * H_ * H_, wbuf, H_ * H_);
    gemm(attb, wbuf, out_b + (size_t)l * H_, tmp, nullptr, M_, H_, H_, 0);
    addln(x, tmp, ln1_g + (size_t)l * H_, ln1_b + (size_t)l * H_);

    conv(fc1_w + (size_t)l * F_ * H_, wbuf, F_ * H_);
    gemm(xb, wbuf, fc1_b + (size_t)l * F_, nullptr, hb, M_, H_, F_, 1);   // GELU
    conv(fc2_w + (size_t)l * H_ * F_, wbuf, H_ * F_);
    gemm(hb, wbuf, fc2_b + (size_t)l * H_, tmp, nullptr, M_, F_, H_, 0);
    addln(x, tmp, ln2_g + (size_t)l * H_, ln2_b + (size_t)l * H_);
  }

  addln(x, nullptr, lnf_g, lnf_b);           // final LayerNorm
  conv(dep_w, wbuf, V_ * H_);
  gemm(xb, wbuf, dep_b, logits, nullptr, M_, H_, V_, 0);   // [B,N,V] logits
}
